// SimMask_42219528520119
// MI455X (gfx1250) — matrix operand order repaired and checked by reading, not compiled, not run on hardware
//
#include <hip/hip_runtime.h>
#include <hip/hip_bf16.h>
#include <math.h>

// ---------------------------------------------------------------------------
// Problem constants (from reference setup_inputs):
//   B=2, N=512, M=H*W=512, C=64, IN_DIM=140, MLP=[128,64], head=[32,2]
// K-permuted layer-1 layout (per-(n,m) channels only; per-n channels folded
// into the bias):
//   k' [0,64)  : dst descriptor (RF3_bnc row)        <- W1 rows 74..137
//   k' 64,65   : dst_xy                              <- W1 rows 5,6
//   k' 66,67   : euc                                 <- W1 rows 7,8
//   k' 68      : euc_norm                            <- W1 row 9
//   k' 69,70   : s1, s2                              <- W1 rows 138,139
//   k' [71,96) : zero padding (K = 3 chunks of 32)
// Folded constants (69): W1 rows 0..4 (xyz_z, src_xy) and 10..73 (src_pts)
// ---------------------------------------------------------------------------
#define BB      2
#define NN      512
#define MM      512
#define CC      64
#define KP      96
#define NFOLD   69
#define H1      128
#define H2      64
#define HHALF   32

typedef __attribute__((ext_vector_type(16))) __bf16 v16bf;
typedef __attribute__((ext_vector_type(8)))  __bf16 v8bf;
typedef __attribute__((ext_vector_type(8)))  float  v8f;

// ---------------------------------------------------------------------------
// WMMA + fragment helpers (CDNA5 wave32, 16x16x32 bf16 -> f32)
// ---------------------------------------------------------------------------
static __device__ inline v8f wmma_bf16(v16bf a, v16bf b, v8f c) {
  return __builtin_amdgcn_wmma_f32_16x16x32_bf16(
      false, a, false, b, (short)0, c, false, false);
}

// A 16x32 bf16 from row-major source, stride ld elements (16B-aligned rows).
// lanes 0-15: row=lane, K = [0..7],[16..23]; lanes 16-31: row=lane-16,
// K = [8..15],[24..31].  Two aligned 16B vector loads per lane.
static __device__ inline v16bf load_a_rowmajor(const __bf16* A, int ld) {
  const int l = threadIdx.x & 31;
  const __bf16* p = A + (l & 15) * ld + ((l & 16) >> 1);
  v8bf lo = *(const v8bf*)p;
  v8bf hi = *(const v8bf*)(p + 16);
  v16bf r;
#pragma unroll
  for (int i = 0; i < 8; ++i) { r[i] = lo[i]; r[i + 8] = hi[i]; }
  return r;
}

// B 32x16 bf16 from row-major W (K x Ncols, stride ld); lane holds column
// col0+(l&15), K = [0..7],[16..23] (lanes 0-15) or [8..15],[24..31].
static __device__ inline v16bf load_b_rowmajor(const __bf16* W, int ld,
                                               int krow0, int col0) {
  const int l = threadIdx.x & 31;
  const __bf16* p = W + (size_t)(krow0 + ((l & 16) >> 1)) * ld + col0 + (l & 15);
  v16bf r;
#pragma unroll
  for (int i = 0; i < 8; ++i) { r[i] = p[(size_t)i * ld]; r[i + 8] = p[(size_t)(i + 16) * ld]; }
  return r;
}

// B 32x16 where logical B[k][j] = Dn[(col0+j)*ld + k] (gather transpose)
static __device__ inline v16bf load_b_colmajor(const __bf16* Dn, int ld,
                                               int k0, int col0) {
  const int l = threadIdx.x & 31;
  const __bf16* p = Dn + (size_t)(col0 + (l & 15)) * ld + k0 + ((l & 16) >> 1);
  v8bf lo = *(const v8bf*)p;
  v8bf hi = *(const v8bf*)(p + 16);
  v16bf r;
#pragma unroll
  for (int i = 0; i < 8; ++i) { r[i] = lo[i]; r[i + 8] = hi[i]; }
  return r;
}

static __device__ inline int ctile_row(int i) {   // C/D tile: VGPR i -> row
  return ((threadIdx.x & 16) ? (i + 8) : i);
}

// ---------------------------------------------------------------------------
// gfx1250 async global->LDS copy (ASYNCcnt-tracked DMA, bypasses VGPRs)
// ---------------------------------------------------------------------------
static __device__ inline void async_copy_b128(const void* gsrc, void* lds_ptr) {
  // low 32 bits of a generic LDS pointer are the LDS byte offset
  unsigned lds_off = (unsigned)(size_t)lds_ptr;
  asm volatile("global_load_async_to_lds_b128 %0, %1, off"
               :: "v"(lds_off), "v"(gsrc)
               : "memory");
}
static __device__ inline void wait_async0() {
  asm volatile("s_wait_asynccnt 0x0" ::: "memory");
}

// ---------------------------------------------------------------------------
// Kernel 0a: pack weights -> bf16.
//   w1b: KP x H1 (permuted/padded W1 rows, per-(n,m) channels)
//   w1c: NFOLD x H1 (W1 rows to be bias-folded: 0..4, 10..73)
//   w2b: H1 x H2
// ---------------------------------------------------------------------------
__global__ __launch_bounds__(256) void prep_w_kernel(
    const float* __restrict__ W1, const float* __restrict__ W2,
    __bf16* __restrict__ w1b, __bf16* __restrict__ w1c,
    __bf16* __restrict__ w2b) {
  int t = blockIdx.x * 256 + threadIdx.x;
  if (t < KP * H1) {
    int k = t / H1, o = t - k * H1;
    int sr = (k < 64) ? (74 + k)
           : (k < 69) ? (5 + (k - 64))
           : (k == 69) ? 138
           : (k == 70) ? 139 : -1;
    w1b[t] = (sr >= 0) ? (__bf16)W1[sr * H1 + o] : (__bf16)0.0f;
  } else if (t < KP * H1 + NFOLD * H1) {
    int u = t - KP * H1;
    int j = u / H1, o = u - j * H1;
    int sr = (j < 5) ? j : (10 + (j - 5));
    w1c[u] = (__bf16)W1[sr * H1 + o];
  } else {
    int u = t - KP * H1 - NFOLD * H1;
    if (u < H1 * H2) w2b[u] = (__bf16)W2[u];
  }
}

// ---------------------------------------------------------------------------
// Kernel 0b: per-row prep (one wave per row).
// ---------------------------------------------------------------------------
__global__ __launch_bounds__(128) void prep_rows_kernel(
    const float* __restrict__ wpts, const float* __restrict__ RF3,
    const float* __restrict__ RF3i, __bf16* __restrict__ srcn,
    __bf16* __restrict__ dstn, __bf16* __restrict__ rf3b,
    float* __restrict__ idxxy) {
  const int gid = blockIdx.x * 4 + (threadIdx.x >> 5);
  const int ln  = threadIdx.x & 31;
  if (gid < BB * NN) {
    const float* p = wpts + (size_t)gid * CC;
    float v0 = p[ln], v1 = p[ln + 32];
    float ss = v0 * v0 + v1 * v1;
#pragma unroll
    for (int o = 16; o > 0; o >>= 1) ss += __shfl_xor(ss, o, 32);
    float d = fmaxf(sqrtf(ss), 1e-12f);
    srcn[(size_t)gid * CC + ln]      = (__bf16)(v0 / d);
    srcn[(size_t)gid * CC + ln + 32] = (__bf16)(v1 / d);
  } else {
    const int i = gid - BB * NN;
    const int b = i >> 9, m = i & 511;
    const float* base = RF3 + (size_t)b * CC * MM + m;
    float v0 = base[(size_t)ln * MM];
    float v1 = base[(size_t)(ln + 32) * MM];
    rf3b[(size_t)i * CC + ln]      = (__bf16)v0;
    rf3b[(size_t)i * CC + ln + 32] = (__bf16)v1;
    float ss = v0 * v0 + v1 * v1;
#pragma unroll
    for (int o = 16; o > 0; o >>= 1) ss += __shfl_xor(ss, o, 32);
    float d = fmaxf(sqrtf(ss), 1e-12f);
    dstn[(size_t)i * CC + ln]      = (__bf16)(v0 / d);
    dstn[(size_t)i * CC + ln + 32] = (__bf16)(v1 / d);
    if (ln < 2)
      idxxy[(size_t)i * 2 + ln] = RF3i[(size_t)b * 3 * MM + (size_t)ln * MM + m];
  }
}

// ---------------------------------------------------------------------------
// Kernel 1: cos[b] = src_n[b] (NxC) @ dst_n[b]^T (CxM) via WMMA bf16.
// ---------------------------------------------------------------------------
__global__ __launch_bounds__(256) void cos_gemm_kernel(
    const __bf16* __restrict__ srcn, const __bf16* __restrict__ dstn,
    float* __restrict__ cosb) {
  const int t  = blockIdx.x * 8 + (threadIdx.x >> 5);
  const int ln = threadIdx.x & 31;
  const int b  = t >> 10;
  const int rm = t & 1023;
  const int nt = rm >> 5;
  const int mt = rm & 31;
  const __bf16* A  = srcn + ((size_t)b * NN + nt * 16) * CC;
  const __bf16* Dn = dstn + (size_t)b * MM * CC;
  v8f acc = {};
#pragma unroll
  for (int kc = 0; kc < 2; ++kc) {
    v16bf af = load_a_rowmajor(A + kc * 32, CC);
    v16bf bf = load_b_colmajor(Dn, CC, kc * 32, mt * 16);
    acc = wmma_bf16(af, bf, acc);
  }
#pragma unroll
  for (int i = 0; i < 8; ++i) {
    int row = nt * 16 + ctile_row(i);
    int col = mt * 16 + (ln & 15);
    cosb[((size_t)b * NN + row) * MM + col] = acc[i];
  }
}

// ---------------------------------------------------------------------------
// Kernel 2: row maxes (over m) and column maxes (over n) of cos.
// ---------------------------------------------------------------------------
__global__ __launch_bounds__(128) void row_col_max_kernel(
    const float* __restrict__ cosb, float* __restrict__ rmax,
    float* __restrict__ cmax) {
  const int gid = blockIdx.x * 4 + (threadIdx.x >> 5);
  const int ln  = threadIdx.x & 31;
  float mx = -1e30f;
  if (gid < BB * NN) {
    const float* p = cosb + (size_t)gid * MM;
    for (int m = ln; m < MM; m += 32) mx = fmaxf(mx, p[m]);
#pragma unroll
    for (int o = 16; o > 0; o >>= 1) mx = fmaxf(mx, __shfl_xor(mx, o, 32));
    if (ln == 0) rmax[gid] = mx;
  } else {
    const int i = gid - BB * NN;
    const int b = i >> 9, m = i & 511;
    const float* p = cosb + (size_t)b * NN * MM + m;
    for (int nn = ln; nn < NN; nn += 32) mx = fmaxf(mx, p[(size_t)nn * MM]);
#pragma unroll
    for (int o = 16; o > 0; o >>= 1) mx = fmaxf(mx, __shfl_xor(mx, o, 32));
    if (ln == 0) cmax[i] = mx;
  }
}

// ---------------------------------------------------------------------------
// Kernel 3 (templated on PHASE): fused MLP over one (b,n) per workgroup.
//  PHASE 1: layer1 pre-act -> BN1 stats
//  PHASE 2: layer1+BN1+ReLU -> layer2 pre-act -> BN2 stats
//  PHASE 3: full chain + online softmax-attention over m -> att (B,N,64)
// ---------------------------------------------------------------------------
template <int PHASE>
__global__ __launch_bounds__(256, 2) void main_pass_kernel(
    const float* __restrict__ wxyz, const float* __restrict__ wpts,
    const float* __restrict__ lz, const float* __restrict__ b1,
    const float* __restrict__ b2, const float* __restrict__ g1,
    const float* __restrict__ be1, const float* __restrict__ g2,
    const float* __restrict__ be2, const __bf16* __restrict__ w1b,
    const __bf16* __restrict__ w1c, const __bf16* __restrict__ w2b,
    const __bf16* __restrict__ rf3b, const float* __restrict__ idxxy,
    const float* __restrict__ cosb, const float* __restrict__ rmax,
    const float* __restrict__ cmax, float* __restrict__ stats1,
    const float* __restrict__ bn1, float* __restrict__ stats2,
    const float* __restrict__ bn2, float* __restrict__ att) {
  __shared__ __bf16 sfeat[16][KP];       // per-(n,m) feature tile
  __shared__ __bf16 sh1[16][H1];         // BN1+ReLU activations
  __shared__ float  sh2[16][H2];         // BN2+ReLU activations (phase 3)
  __shared__ float  sconst[NFOLD];       // bf16-rounded folded features
  __shared__ float  sM4[4], sD4[4], sA4[4][H2];  // online-softmax merge

  const int b   = blockIdx.x >> 9;
  const int n   = blockIdx.x & 511;
  const int tid = threadIdx.x;
  const int wv  = tid >> 5;
  const int ln  = tid & 31;

  __builtin_prefetch(w1b, 0, 3);
  __builtin_prefetch(w2b, 0, 3);

  // per-(b,n) scalars
  const float ax  = wxyz[((size_t)(b * NN + n)) * 3 + 0];
  const float ay  = wxyz[((size_t)(b * NN + n)) * 3 + 1];
  const float az  = wxyz[((size_t)(b * NN + n)) * 3 + 2];
  const float lzv = lz[b * NN + n];
  const float rmx = rmax[b * NN + n];

  // init: zero sfeat (pad region stays zero), fill folded-constant vector
  for (int e = tid; e < 16 * KP; e += 256) ((__bf16*)sfeat)[e] = (__bf16)0.0f;
  if (tid < NFOLD) {
    float v;
    if (tid < 3)       v = wxyz[((size_t)(b * NN + n)) * 3 + tid] * lzv;
    else if (tid < 5)  v = wpts[((size_t)(b * NN + n)) * CC + (tid - 3)];
    else               v = wpts[((size_t)(b * NN + n)) * CC + (tid - 5)];
    sconst[tid] = (float)(__bf16)v;      // match bf16 feature rounding
  }
  __syncthreads();

  // layer-1 channel parameters for this wave's column tile
  const int ch1 = wv * 16 + (ln & 15);
  float bias1 = b1[ch1];
  for (int j = 0; j < NFOLD; ++j)        // fold per-n channels into bias
    bias1 += sconst[j] * (float)w1c[(size_t)j * H1 + ch1];
  float mean1v = 0.f, is1v = 1.f, gg1 = 1.f, bb1 = 0.f;
  if (PHASE >= 2) { mean1v = bn1[ch1]; is1v = bn1[H1 + ch1]; gg1 = g1[ch1]; bb1 = be1[ch1]; }
  // layer-2 parameters (waves 0..3 compute; loaded uniformly)
  const int ch2 = (wv & 3) * 16 + (ln & 15);
  const float bias2 = b2[ch2];
  float mean2v = 0.f, is2v = 1.f, gg2 = 1.f, bb2 = 0.f;
  if (PHASE >= 3) { mean2v = bn2[ch2]; is2v = bn2[H2 + ch2]; gg2 = g2[ch2]; bb2 = be2[ch2]; }

  // hoist weight B-fragments (loop invariant)
  v16bf w1f[3];
#pragma unroll
  for (int kc = 0; kc < 3; ++kc) w1f[kc] = load_b_rowmajor(w1b, H1, kc * 32, wv * 16);
  v16bf w2f[4];
#pragma unroll
  for (int kc = 0; kc < 4; ++kc) w2f[kc] = load_b_rowmajor(w2b, H2, kc * 32, (wv & 3) * 16);

  float acc_s1 = 0.f, acc_q1 = 0.f;      // BN1 stats (phase 1)
  float acc_s2 = 0.f, acc_q2 = 0.f;      // BN2 stats (phase 2)
  float runM = -1e30f, runD = 0.f;       // online softmax (phase 3, waves 0..3)
  float at0 = 0.f, at1 = 0.f;

  for (int mt = 0; mt < MM / 16; ++mt) {
    __syncthreads();
    // ---- dst-descriptor tile: async DMA global -> LDS (waves 0..3) ---------
    if (tid < 128) {
      const int r = tid >> 3, seg = tid & 7;
      const int m = mt * 16 + r;
      async_copy_b128(rf3b + ((size_t)(b * MM + m)) * CC + seg * 8,
                      &sfeat[r][seg * 8]);
    }
    // ---- per-(n,m) geom + sim channels (threads 128..239) ------------------
    if (tid >= 128 && tid < 128 + 16 * 7) {
      const int e = tid - 128;
      const int r = e / 7;
      const int c = 64 + (e - r * 7);
      const int m = mt * 16 + r;
      const float ix = idxxy[(size_t)(b * MM + m) * 2 + 0];
      const float iy = idxxy[(size_t)(b * MM + m) * 2 + 1];
      float v;
      if (c == 64)      v = ix;
      else if (c == 65) v = iy;
      else if (c == 66) v = ax - ix;
      else if (c == 67) v = ay - iy;
      else if (c == 68) {
        float dx = ax - ix, dy = ay - iy;
        v = sqrtf(dx * dx + dy * dy);
      } else {
        float cv = cosb[((size_t)(b * NN + n)) * MM + m];
        v = (c == 69) ? cv / (rmx + 1e-6f)
                      : cv / (cmax[b * MM + m] + 1e-10f);
      }
      sfeat[r][c] = (__bf16)v;
    }
    if (wv < 4) wait_async0();           // DMA done before the barrier
    __syncthreads();

    // ---- layer 1: (16 x 96) @ (96 x 128); wave wv owns cols [16wv,16wv+16)
    v8f acc = {};
#pragma unroll
    for (int kc = 0; kc < 3; ++kc) {
      v16bf af = load_a_rowmajor(&sfeat[0][kc * 32], KP);
      acc = wmma_bf16(af, w1f[kc], acc);
    }
#pragma unroll
    for (int i = 0; i < 8; ++i) acc[i] += bias1;

    if (PHASE == 1) {
#pragma unroll
      for (int i = 0; i < 8; ++i) { acc_s1 += acc[i]; acc_q1 += acc[i] * acc[i]; }
      continue;                           // uniform across block
    }

    // ---- BN1 + ReLU -> sh1 (bf16) -----------------------------------------
#pragma unroll
    for (int i = 0; i < 8; ++i) {
      float y = (acc[i] - mean1v) * is1v * gg1 + bb1;
      sh1[ctile_row(i)][ch1] = (__bf16)fmaxf(y, 0.0f);
    }
    __syncthreads();

    // ---- layer 2: (16 x 128) @ (128 x 64); waves 0..3 ----------------------
    if (wv < 4) {
      v8f acc2 = {};
#pragma unroll
      for (int kc = 0; kc < 4; ++kc) {
        v16bf af = load_a_rowmajor(&sh1[0][kc * 32], H1);
        acc2 = wmma_bf16(af, w2f[kc], acc2);
      }
#pragma unroll
      for (int i = 0; i < 8; ++i) acc2[i] += bias2;
      if (PHASE == 2) {
#pragma unroll
        for (int i = 0; i < 8; ++i) { acc_s2 += acc2[i]; acc_q2 += acc2[i] * acc2[i]; }
      }
      if (PHASE == 3) {
#pragma unroll
        for (int i = 0; i < 8; ++i) {
          float y = (acc2[i] - mean2v) * is2v * gg2 + bb2;
          sh2[ctile_row(i)][ch2] = fmaxf(y, 0.0f);
        }
      }
    }
    if (PHASE == 2) continue;             // uniform across block

    __syncthreads();
    // ---- phase 3: online softmax-attention (waves 0..3, 4 rows each) -------
    if (wv < 4) {
      for (int r = wv * 4; r < wv * 4 + 4; ++r) {
        float x0 = sh2[r][ln], x1 = sh2[r][ln + 32];
        float lm = fmaxf(x0, x1);
#pragma unroll
        for (int o = 16; o > 0; o >>= 1) lm = fmaxf(lm, __shfl_xor(lm, o, 32));
        float nM = fmaxf(runM, lm);
        float sc = __expf(runM - nM);
        float ee = __expf(lm - nM);
        runD = runD * sc + ee;
        at0  = at0 * sc + ee * x0;
        at1  = at1 * sc + ee * x1;
        runM = nM;
      }
    }
  }

  if (PHASE == 1) {
    atomicAdd(&stats1[ch1], acc_s1);
    atomicAdd(&stats1[H1 + ch1], acc_q1);
  }
  if (PHASE == 2 && wv < 4) {
    atomicAdd(&stats2[ch2], acc_s2);
    atomicAdd(&stats2[H2 + ch2], acc_q2);
  }
  if (PHASE == 3) {
    if (wv < 4) {                         // publish per-wave softmax state
      if (ln == 0) { sM4[wv] = runM; sD4[wv] = runD; }
      sA4[wv][ln]      = at0;
      sA4[wv][ln + 32] = at1;
    }
    __syncthreads();
    if (wv == 0) {                        // merge 4 online states
      float Mg = fmaxf(fmaxf(sM4[0], sM4[1]), fmaxf(sM4[2], sM4[3]));
      float D = 0.f, a0 = 0.f, a1 = 0.f;
#pragma unroll
      for (int w = 0; w < 4; ++w) {
        float sc = __expf(sM4[w] - Mg);
        D  += sc * sD4[w];
        a0 += sc * sA4[w][ln];
        a1 += sc * sA4[w][ln + 32];
      }
      float inv = 1.0f / D;
      att[((size_t)(b * NN + n)) * CC + ln]      = a0 * inv;
      att[((size_t)(b * NN + n)) * CC + ln + 32] = a1 * inv;
    }
  }
}

// ---------------------------------------------------------------------------
// BN finalize: mean = sum/cnt; invstd = rsqrt(sumsq/cnt - mean^2 + 1e-5)
// ---------------------------------------------------------------------------
__global__ __launch_bounds__(128) void finalize_bn_kernel(
    const float* __restrict__ stats, float* __restrict__ bn, int C, float cnt) {
  int t = threadIdx.x;
  if (t < C) {
    float mean = stats[t] / cnt;
    float var  = stats[C + t] / cnt - mean * mean;
    bn[t]     = mean;
    bn[C + t] = rsqrtf(var + 1e-5f);
  }
}

// ---------------------------------------------------------------------------
// Head kernel 1: z = att @ M1w + M1b (1024 x 64 -> 32) + BN stats
// ---------------------------------------------------------------------------
__global__ __launch_bounds__(256) void head1_kernel(
    const float* __restrict__ att, const float* __restrict__ M1w,
    const float* __restrict__ M1b, float* __restrict__ zbuf,
    float* __restrict__ stats3) {
  int t = blockIdx.x * 256 + threadIdx.x;
  int row = t >> 5, o = t & 31;
  float s = 0.0f;
  for (int c = 0; c < CC; ++c) s += att[(size_t)row * CC + c] * M1w[c * HHALF + o];
  s += M1b[o];
  zbuf[t] = s;
  atomicAdd(&stats3[o], s);
  atomicAdd(&stats3[HHALF + o], s * s);
}

// ---------------------------------------------------------------------------
// Head kernel 2: logits = relu(BN(z)) @ M2w + M2b (1024 x 32 -> 2)
// ---------------------------------------------------------------------------
__global__ __launch_bounds__(256) void head2_kernel(
    const float* __restrict__ zbuf, const float* __restrict__ bn3,
    const float* __restrict__ M1g, const float* __restrict__ M1be,
    const float* __restrict__ M2w, const float* __restrict__ M2b,
    float* __restrict__ out) {
  int row = blockIdx.x * 256 + threadIdx.x;
  if (row >= BB * NN) return;
  float o0 = M2b[0], o1 = M2b[1];
  for (int c = 0; c < HHALF; ++c) {
    float z = zbuf[(size_t)row * HHALF + c];
    float y = (z - bn3[c]) * bn3[HHALF + c] * M1g[c] + M1be[c];
    y = fmaxf(y, 0.0f);
    o0 += y * M2w[c * 2 + 0];
    o1 += y * M2w[c * 2 + 1];
  }
  out[row * 2 + 0] = o0;
  out[row * 2 + 1] = o1;
}

// ---------------------------------------------------------------------------
// Launch
// ---------------------------------------------------------------------------
extern "C" void kernel_launch(void* const* d_in, const int* in_sizes, int n_in,
                              void* d_out, int out_size, void* d_ws, size_t ws_size,
                              hipStream_t stream) {
  const float* wxyz = (const float*)d_in[0];
  const float* wpts = (const float*)d_in[1];
  const float* RF3  = (const float*)d_in[2];
  const float* RF3i = (const float*)d_in[3];
  const float* lz   = (const float*)d_in[4];
  const float* W1   = (const float*)d_in[5];
  const float* b1   = (const float*)d_in[6];
  const float* g1   = (const float*)d_in[7];
  const float* be1  = (const float*)d_in[8];
  const float* W2   = (const float*)d_in[9];
  const float* b2   = (const float*)d_in[10];
  const float* g2   = (const float*)d_in[11];
  const float* be2  = (const float*)d_in[12];
  const float* M1w  = (const float*)d_in[13];
  const float* M1b  = (const float*)d_in[14];
  const float* M1g  = (const float*)d_in[15];
  const float* M1be = (const float*)d_in[16];
  const float* M2w  = (const float*)d_in[17];
  const float* M2b  = (const float*)d_in[18];
  float* out = (float*)d_out;

  char* ws = (char*)d_ws;
  size_t off = 0;
  auto alloc = [&](size_t bytes) -> void* {
    void* p = ws + off;
    off = (off + bytes + 255) & ~(size_t)255;
    return p;
  };
  __bf16* srcn  = (__bf16*)alloc((size_t)BB * NN * CC * 2);
  __bf16* dstn  = (__bf16*)alloc((size_t)BB * MM * CC * 2);
  __bf16* rf3b  = (__bf16*)alloc((size_t)BB * MM * CC * 2);
  __bf16* w1b   = (__bf16*)alloc((size_t)KP * H1 * 2);
  __bf16* w1c   = (__bf16*)alloc((size_t)NFOLD * H1 * 2);
  __bf16* w2b   = (__bf16*)alloc((size_t)H1 * H2 * 2);
  float*  idxxy = (float*)alloc((size_t)BB * MM * 2 * 4);
  float*  cosb  = (float*)alloc((size_t)BB * NN * MM * 4);
  float*  rmax  = (float*)alloc((size_t)BB * NN * 4);
  float*  cmax  = (float*)alloc((size_t)BB * MM * 4);
  float*  stats1 = (float*)alloc(2 * H1 * 4);
  float*  bn1    = (float*)alloc(2 * H1 * 4);
  float*  stats2 = (float*)alloc(2 * H2 * 4);
  float*  bn2    = (float*)alloc(2 * H2 * 4);
  float*  attb   = (float*)alloc((size_t)BB * NN * CC * 4);
  float*  zbuf   = (float*)alloc((size_t)BB * NN * HHALF * 4);
  float*  stats3 = (float*)alloc(2 * HHALF * 4);
  float*  bn3    = (float*)alloc(2 * HHALF * 4);
  (void)in_sizes; (void)n_in; (void)out_size; (void)ws_size;

  // zero atomic-accumulated stats each launch (deterministic, capture-safe)
  hipMemsetAsync(stats1, 0, 2 * H1 * 4, stream);
  hipMemsetAsync(stats2, 0, 2 * H2 * 4, stream);
  hipMemsetAsync(stats3, 0, 2 * HHALF * 4, stream);

  // prep
  const int wtot = KP * H1 + NFOLD * H1 + H1 * H2;
  prep_w_kernel<<<(wtot + 255) / 256, 256, 0, stream>>>(W1, W2, w1b, w1c, w2b);
  prep_rows_kernel<<<(BB * (NN + MM)) / 4, 128, 0, stream>>>(wpts, RF3, RF3i, srcn, dstn, rf3b, idxxy);

  // cosine similarity GEMM + maxes
  cos_gemm_kernel<<<(BB * (NN / 16) * (MM / 16)) / 8, 256, 0, stream>>>(srcn, dstn, cosb);
  row_col_max_kernel<<<(BB * (NN + MM)) / 4, 128, 0, stream>>>(cosb, rmax, cmax);

  const float cnt12 = (float)((size_t)BB * NN * MM);
  const dim3 mgrid(BB * NN);

  // phase 1: BN1 stats
  main_pass_kernel<1><<<mgrid, 256, 0, stream>>>(
      wxyz, wpts, lz, b1, b2, g1, be1, g2, be2, w1b, w1c, w2b, rf3b, idxxy,
      cosb, rmax, cmax, stats1, bn1, stats2, bn2, attb);
  finalize_bn_kernel<<<1, 128, 0, stream>>>(stats1, bn1, H1, cnt12);

  // phase 2: BN2 stats
  main_pass_kernel<2><<<mgrid, 256, 0, stream>>>(
      wxyz, wpts, lz, b1, b2, g1, be1, g2, be2, w1b, w1c, w2b, rf3b, idxxy,
      cosb, rmax, cmax, stats1, bn1, stats2, bn2, attb);
  finalize_bn_kernel<<<1, 128, 0, stream>>>(stats2, bn2, H2, cnt12);

  // phase 3: full chain + attention
  main_pass_kernel<3><<<mgrid, 256, 0, stream>>>(
      wxyz, wpts, lz, b1, b2, g1, be1, g2, be2, w1b, w1c, w2b, rf3b, idxxy,
      cosb, rmax, cmax, stats1, bn1, stats2, bn2, attb);

  // head MLP: 64 -> 32 (BN+ReLU) -> 2
  head1_kernel<<<(BB * NN * HHALF) / 256, 256, 0, stream>>>(attb, M1w, M1b, zbuf, stats3);
  finalize_bn_kernel<<<1, 128, 0, stream>>>(stats3, bn3, HHALF, (float)(BB * NN));
  head2_kernel<<<(BB * NN + 255) / 256, 256, 0, stream>>>(zbuf, bn3, M1g, M1be, M2w, M2b, out);
}
